// SAGE_3186865734220
// MI455X (gfx1250) — matrix operand order repaired and checked by reading, not compiled, not run on hardware
//
#include <hip/hip_runtime.h>
#include <hip/hip_bf16.h>

// ---------------------------------------------------------------------------
// GraphSAGE 2-layer forward for MI455X (gfx1250, wave32).
//   scatter    : per-edge atomic accumulation of neighbor features (+degree)
//   pack_wfrag : one-shot pre-pack of weights into WMMA B-fragment layout
//   fused GEMM : mean-divide + [mean@Wl^T + x@Wr^T + b] via
//                v_wmma_f32_16x16x32_bf16 (f32 accumulate), templated on NOUT
// ---------------------------------------------------------------------------

typedef __attribute__((ext_vector_type(16))) __bf16 v16bf;
typedef __attribute__((ext_vector_type(8)))  float  v8f;
typedef __attribute__((ext_vector_type(2)))  float  v2f;   // builtin-compatible

union AFrag {
    v16bf        v;
    unsigned int u[8];
    uint4        q[2];
};

__device__ __forceinline__ unsigned int packbf(float a, float b) {
    // native bf16 converts -> packed v_cvt on gfx1250
    union { __bf16 h[2]; unsigned int u; } p;
    p.h[0] = (__bf16)a;
    p.h[1] = (__bf16)b;
    return p.u;
}

__global__ __launch_bounds__(256) void zero_f32v4(float4* __restrict__ p, int n4) {
    int i = blockIdx.x * blockDim.x + threadIdx.x;
    if (i < n4) p[i] = make_float4(0.f, 0.f, 0.f, 0.f);
}

// One (edge, channel) pair per thread; a wave covers 32 contiguous channels of
// one edge: coalesced 128B gather and a coalesced atomic stream into L2.
__global__ __launch_bounds__(256) void scatter_accum(
    const float* __restrict__ feat,
    const int*   __restrict__ src,
    const int*   __restrict__ dst,
    float* __restrict__ agg,
    float* __restrict__ cnt,
    int nEdges)
{
    int t = blockIdx.x * blockDim.x + threadIdx.x;
    int e = t >> 6;
    int c = t & 63;
    if (e >= nEdges) return;
    if (c == 0) {
        // keep the edge-index stream ahead of us (global_prefetch_b8)
        __builtin_prefetch(&src[e + 4096], 0, 0);
        __builtin_prefetch(&dst[e + 4096], 0, 0);
    }
    int s = src[e];
    int d = dst[e];
    float v = feat[(size_t)s * 64 + c];
    atomicAdd(&agg[(size_t)d * 64 + c], v);
    if (c == 0) atomicAdd(&cnt[d], 1.0f);
}

// Pre-pack W [nOut,64] row-major f32 into WMMA B-fragment lane layout:
//   out[((colTile*2 + kf)*32 + lane)*8 + j] = bf16pair( B[k][n], B[k+1][n] )
//   with n = colTile*16 + (lane&15), k = kf*32 + (lane>>4)*8 + (j<4 ? 2j : 16+2(j-4)),
//   B[k][n] = W[n][k].
__global__ __launch_bounds__(256) void pack_wfrag(
    const float* __restrict__ W, unsigned int* __restrict__ out, int total)
{
    int idx = blockIdx.x * blockDim.x + threadIdx.x;
    if (idx >= total) return;
    const int K = 64;
    int j       = idx & 7;
    int lane    = (idx >> 3) & 31;
    int kf      = (idx >> 8) & 1;
    int colTile = idx >> 9;
    int ncol    = colTile * 16 + (lane & 15);
    int k       = kf * 32 + (lane >> 4) * 8 + ((j < 4) ? (2 * j) : (16 + 2 * (j - 4)));
    out[idx] = packbf(W[(size_t)ncol * K + k], W[(size_t)ncol * K + k + 1]);
}

// Fused: mean = agg / max(cnt,1); out = mean @ Wl^T + b + x @ Wr^T; opt. ReLU.
// K = 64 fixed. 256 threads = 8 waves; each wave owns one 16x16 output tile
// (tilesN = NOUT/16 column tiles, 8/tilesN row sub-tiles).
template <int NOUT, bool RELU>
__global__ __launch_bounds__(256) void sage_fused_gemm(
    const float* __restrict__ agg,
    const float* __restrict__ cnt,
    const float* __restrict__ xin,
    const unsigned int* __restrict__ preL,   // packed B fragments of Wl
    const unsigned int* __restrict__ preR,   // packed B fragments of Wr
    const float* __restrict__ bl,            // [NOUT]
    float* __restrict__ out,                 // [nNodes, NOUT]
    int nNodes)
{
    constexpr int K            = 64;
    constexpr int tilesN       = NOUT / 16;           // 4 (layer1) or 2 (layer2)
    constexpr int rowsPerBlock = (8 / tilesN) * 16;   // 32 or 64
    const int rowBase = blockIdx.x * rowsPerBlock;

    __shared__ __bf16 sM[64][68];   // bf16 mean, padded rows (bank spread)
    __shared__ __bf16 sX[64][68];   // bf16 x
    __shared__ float  sInv[64];

    const int tid = threadIdx.x;

    if (tid < rowsPerBlock) {
        int node = rowBase + tid;
        float c = (node < nNodes) ? cnt[node] : 1.0f;
        sInv[tid] = 1.0f / fmaxf(c, 1.0f);
    }
    __syncthreads();

    // Coalesced stage of activations -> LDS as packed bf16 pairs (ds_store_b32)
    for (int i = tid; i < rowsPerBlock * 32; i += 256) {
        int r = i >> 5, cp = (i & 31) << 1;
        int node = rowBase + r;
        float m0 = 0.f, m1 = 0.f, x0 = 0.f, x1 = 0.f;
        if (node < nNodes) {
            float inv = sInv[r];
            v2f av = __builtin_nontemporal_load(
                (const v2f*)&agg[(size_t)node * K + cp]);   // agg is dead after this
            v2f xv = *(const v2f*)&xin[(size_t)node * K + cp];
            m0 = av.x * inv;  m1 = av.y * inv;
            x0 = xv.x;        x1 = xv.y;
        }
        *(unsigned int*)&sM[r][cp] = packbf(m0, m1);
        *(unsigned int*)&sX[r][cp] = packbf(x0, x1);
    }
    __syncthreads();

    const int wave    = tid >> 5;
    const int lane    = tid & 31;
    const int colTile = wave % tilesN;
    const int rowSub  = wave / tilesN;
    const int rBase   = rowSub << 4;
    const int half    = lane >> 4;   // 0: lanes 0-15, 1: lanes 16-31
    const int l15     = lane & 15;

    // A fragments (16x32 bf16): lane holds row M=l15; VGPR j holds K pair
    //   j<4 : K = 2j       + 8*half (+K0)
    //   j>=4: K = 16+2(j-4)+ 8*half (+K0)
    AFrag aM[2], aX[2];
#pragma unroll
    for (int kf = 0; kf < 2; ++kf) {
        const int K0 = kf * 32;
#pragma unroll
        for (int j = 0; j < 8; ++j) {
            int k = K0 + (half ? 8 : 0) + ((j < 4) ? (2 * j) : (16 + 2 * (j - 4)));
            aM[kf].u[j] = *(const unsigned int*)&sM[rBase + l15][k];
            aX[kf].u[j] = *(const unsigned int*)&sX[rBase + l15][k];
        }
    }

    // B fragments: two contiguous 16B loads per fragment from the pre-packed
    // tables (global_load_b128, L2-resident).
    AFrag bL[2], bR[2];
#pragma unroll
    for (int kf = 0; kf < 2; ++kf) {
        const uint4* pL = (const uint4*)(preL + ((size_t)((colTile * 2 + kf) * 32 + lane)) * 8);
        const uint4* pR = (const uint4*)(preR + ((size_t)((colTile * 2 + kf) * 32 + lane)) * 8);
        bL[kf].q[0] = pL[0];  bL[kf].q[1] = pL[1];
        bR[kf].q[0] = pR[0];  bR[kf].q[1] = pR[1];
    }

    // C init with bias (broadcast down the M dimension of the tile)
    const int   ncol = colTile * 16 + l15;
    const float bv   = bl[ncol];
    v8f c = {bv, bv, bv, bv, bv, bv, bv, bv};

    c = __builtin_amdgcn_wmma_f32_16x16x32_bf16(false, aM[0].v, false, bL[0].v,
                                                (short)0, c, false, false);
    c = __builtin_amdgcn_wmma_f32_16x16x32_bf16(false, aM[1].v, false, bL[1].v,
                                                (short)0, c, false, false);
    c = __builtin_amdgcn_wmma_f32_16x16x32_bf16(false, aX[0].v, false, bR[0].v,
                                                (short)0, c, false, false);
    c = __builtin_amdgcn_wmma_f32_16x16x32_bf16(false, aX[1].v, false, bR[1].v,
                                                (short)0, c, false, false);

    if (RELU) {
#pragma unroll
        for (int r = 0; r < 8; ++r) c[r] = fmaxf(c[r], 0.0f);
    }

    // D layout: VGPR r -> M = r + 8*half, N = l15.
    // Single base pointer + immediate offsets (NOUT is a compile-time const);
    // wave-uniform fast path when the whole sub-tile is in range.
    const int nodeBase = rowBase + rBase + half * 8;
    float* po = out + (size_t)nodeBase * NOUT + ncol;
    if (nodeBase + 8 <= nNodes) {
#pragma unroll
        for (int r = 0; r < 8; ++r) {
            if (RELU) po[r * NOUT] = c[r];
            else      __builtin_nontemporal_store(c[r], &po[r * NOUT]);
        }
    } else {
#pragma unroll
        for (int r = 0; r < 8; ++r) {
            if (nodeBase + r < nNodes) {
                if (RELU) po[r * NOUT] = c[r];
                else      __builtin_nontemporal_store(c[r], &po[r * NOUT]);
            }
        }
    }
}

extern "C" void kernel_launch(void* const* d_in, const int* in_sizes, int n_in,
                              void* d_out, int out_size, void* d_ws, size_t ws_size,
                              hipStream_t stream) {
    const float* x   = (const float*)d_in[0];
    const int*   ei  = (const int*)d_in[1];
    const float* W1l = (const float*)d_in[2];
    const float* b1l = (const float*)d_in[3];
    const float* W1r = (const float*)d_in[4];
    const float* W2l = (const float*)d_in[5];
    const float* b2l = (const float*)d_in[6];
    const float* W2r = (const float*)d_in[7];

    const int C = 64;
    const int N = in_sizes[0] / C;       // 100000 nodes
    const int E = in_sizes[1] / 2;       // 1000000 edges
    const int* src = ei;
    const int* dst = ei + E;

    float* agg = (float*)d_ws;                 // [N,64]
    float* cnt = agg + (size_t)N * C;          // [N]   (contiguous after agg)
    float* h   = cnt + N;                      // [N,64]
    unsigned int* preL1 = (unsigned int*)(h + (size_t)N * C);  // 4*2*32*8 = 2048
    unsigned int* preR1 = preL1 + 2048;
    unsigned int* preL2 = preR1 + 2048;        // 2*2*32*8 = 1024
    unsigned int* preR2 = preL2 + 1024;

    const int zeroN   = N * C + N;             // agg + cnt in one shot
    const int zeroN4  = zeroN >> 2;            // 65N divisible by 4 (N=100000)
    const int zeroBlk = (zeroN4 + 255) / 256;
    const int scatBlk = (E * C + 255) / 256;

    // One-shot weight fragment packing (tiny; L2-resident afterwards)
    pack_wfrag<<<8, 256, 0, stream>>>(W1l, preL1, 2048);
    pack_wfrag<<<8, 256, 0, stream>>>(W1r, preR1, 2048);
    pack_wfrag<<<4, 256, 0, stream>>>(W2l, preL2, 1024);
    pack_wfrag<<<4, 256, 0, stream>>>(W2r, preR2, 1024);

    // ---- Layer 1 ----
    zero_f32v4<<<zeroBlk, 256, 0, stream>>>((float4*)agg, zeroN4);
    scatter_accum<<<scatBlk, 256, 0, stream>>>(x, src, dst, agg, cnt, E);
    {
        const int rowsPerBlock = 32;           // nOut=64 -> 4 col x 2 row tiles
        const int grid = (N + rowsPerBlock - 1) / rowsPerBlock;
        sage_fused_gemm<64, true><<<grid, 256, 0, stream>>>(agg, cnt, x,
                                                            preL1, preR1, b1l,
                                                            h, N);
    }

    // ---- Layer 2 ----
    zero_f32v4<<<zeroBlk, 256, 0, stream>>>((float4*)agg, zeroN4);
    scatter_accum<<<scatBlk, 256, 0, stream>>>(h, src, dst, agg, cnt, E);
    {
        const int rowsPerBlock = 64;           // nOut=32 -> 2 col x 4 row tiles
        const int grid = (N + rowsPerBlock - 1) / rowsPerBlock;
        sage_fused_gemm<32, false><<<grid, 256, 0, stream>>>(agg, cnt, h,
                                                             preL2, preR2, b2l,
                                                             (float*)d_out, N);
    }
}
